// BlockwiseEarlyExitMamba_44143673868461
// MI455X (gfx1250) — matrix operand order repaired and checked by reading, not compiled, not run on hardware
//
#include <hip/hip_runtime.h>
#include <hip/hip_bf16.h>

// ---------------------------------------------------------------------------
// BlockwiseEarlyExitMamba for MI455X (gfx1250, wave32, WMMA)
//
// Early-exit insight: output = f(feat[:, 31, :]) and the whole network is
// causal, so only the first L_EFF=32 of 64 timesteps are ever computed.
// Token-parallel GEMMs run on v_wmma_f32_16x16x32_bf16 (f32 accum) with
// fragments loaded straight from global memory (L2-resident working set,
// no LDS round trip); the sequential scan / conv / LN stay in f32 VALU.
// ---------------------------------------------------------------------------

typedef __bf16 bf16n;
typedef __attribute__((ext_vector_type(16))) __bf16 v16bf;
typedef __attribute__((ext_vector_type(8)))  float  v8f;

#define D_MODEL 256
#define D_INNER 512
#define D_STATE 16
#define D_CONV  4
#define DT_RANK 16
#define N_LAYERS 4
#define BATCH   128
#define L_IN    64
#define L_EFF   32            // EXIT_POS; causal truncation
#define NTOK    (BATCH * L_EFF)   // 4096 tokens
#define LN_EPS  1e-5f

// ---------------------------------------------------------------------------
// f32 -> bf16 weight conversion (once per call, into workspace)
// ---------------------------------------------------------------------------
__global__ void f2bf_kernel(const float* __restrict__ src, bf16n* __restrict__ dst, int n) {
    int i = blockIdx.x * blockDim.x + threadIdx.x;
    if (i < n) dst[i] = (bf16n)src[i];
}

// ---------------------------------------------------------------------------
// Tokenizer: embeddings + scalar projections -> 136-d concat -> fusion GEMV
// (256x136 per token) -> LayerNorm.  One block (256 threads) per token.
// ---------------------------------------------------------------------------
__global__ void tok_kernel(const float* __restrict__ x,
                           const float* __restrict__ emb_proto,
                           const float* __restrict__ emb_flags,
                           const float* __restrict__ emb_dir,
                           const float* __restrict__ plw, const float* __restrict__ plb,
                           const float* __restrict__ piw, const float* __restrict__ pib,
                           const float* __restrict__ fw,  const float* __restrict__ fb,
                           const float* __restrict__ ng,  const float* __restrict__ nb,
                           float* __restrict__ feat, bf16n* __restrict__ featb) {
    __shared__ float cat[136];
    __shared__ float red[256];
    const int tok = blockIdx.x;
    const int b = tok / L_EFF, l = tok % L_EFF;
    const int t = threadIdx.x;
    const float* xr = x + (size_t)(b * L_IN + l) * 5;

    if (t < 136) {
        float v;
        if (t < 32) {
            int proto = (int)xr[0]; proto = min(max(proto, 0), 255);
            v = emb_proto[proto * 32 + t];
        } else if (t < 64) {
            v = xr[1] * plw[t - 32] + plb[t - 32];
        } else if (t < 96) {
            int flags = (int)xr[2]; flags = min(max(flags, 0), 63);
            v = emb_flags[flags * 32 + (t - 64)];
        } else if (t < 128) {
            v = xr[3] * piw[t - 96] + pib[t - 96];
        } else {
            int dir = (int)xr[4]; dir = min(max(dir, 0), 1);
            v = emb_dir[dir * 8 + (t - 128)];
        }
        cat[t] = v;
    }
    __syncthreads();

    float s = fb[t];
    #pragma unroll 8
    for (int k = 0; k < 136; ++k) s = fmaf(cat[k], fw[t * 136 + k], s);

    // LayerNorm over 256 channels
    red[t] = s; __syncthreads();
    for (int st = 128; st > 0; st >>= 1) { if (t < st) red[t] += red[t + st]; __syncthreads(); }
    const float mu = red[0] * (1.f / D_MODEL); __syncthreads();
    const float d = s - mu;
    red[t] = d * d; __syncthreads();
    for (int st = 128; st > 0; st >>= 1) { if (t < st) red[t] += red[t + st]; __syncthreads(); }
    const float var = red[0] * (1.f / D_MODEL);
    const float o = d * rsqrtf(var + LN_EPS) * ng[t] + nb[t];
    feat [(size_t)tok * D_MODEL + t] = o;
    featb[(size_t)tok * D_MODEL + t] = (bf16n)o;
}

// ---------------------------------------------------------------------------
// bf16 WMMA GEMM: C[M,N] (f32) = A[M,K] (bf16, row-major) * W[N,K]^T (bf16).
// One wave32 per 16M x (16*NT)N stripe.  Key observation: for the CDNA5
// 16x16x32 bf16 fragment layouts (CDNA5 ISA 7.12.2) and a
// row-major source, each lane's A fragment is two contiguous 16B chunks of
// one row, and each lane's B fragment is likewise two 16B chunks of one
// weight row -> load fragments directly with global_load_b128.  No LDS, no
// barriers; A fragment reused across NT weight tiles.
// ---------------------------------------------------------------------------
template <int NT>
__global__ void __launch_bounds__(32)
gemm_bf16_kernel(const bf16n* __restrict__ A, const bf16n* __restrict__ W,
                 float* __restrict__ C, int M, int N, int K) {
    const int lane = threadIdx.x;          // 0..31
    const int m0 = blockIdx.y * 16;
    const int n0 = blockIdx.x * (16 * NT);
    const int rr = lane & 15;              // row (A) / col (B) owned by lane
    const int hl = lane >> 4;              // lane half

    const uint4* rowA = (const uint4*)(A + (size_t)(m0 + rr) * K);
    const uint4* rowB[NT];
    #pragma unroll
    for (int t = 0; t < NT; ++t)
        rowB[t] = (const uint4*)(W + (size_t)(n0 + t * 16 + rr) * K);

    v8f acc[NT] = {};

    const int ksteps = K >> 3;             // uint4 (8 bf16) granules per row
    for (int k8 = 0; k8 < ksteps; k8 += 4) {   // 32 K-elems per iteration
        // A 16x32 fragment: af[0..7] = A[m][k0 + hl*8 .. +8],
        //                   af[8..15] = A[m][k0+16+hl*8 .. +8]
        union { uint4 q[2]; v16bf v; } fa;
        fa.q[0] = rowA[k8 + hl];
        fa.q[1] = rowA[k8 + hl + 2];
        #pragma unroll
        for (int t = 0; t < NT; ++t) {
            // B 32x16 fragment: bf[0..7] = W[n][k0 + hl*8 .. +8],
            //                   bf[8..15] = W[n][k0+16+hl*8 .. +8]
            union { uint4 q[2]; v16bf v; } fb_;
            fb_.q[0] = rowB[t][k8 + hl];
            fb_.q[1] = rowB[t][k8 + hl + 2];
            acc[t] = __builtin_amdgcn_wmma_f32_16x16x32_bf16(
                         false, fa.v, false, fb_.v, (short)0, acc[t], false, false);
        }
    }
    // D layout: lane rr = N, VGPR r -> M = r + 8*half
    #pragma unroll
    for (int t = 0; t < NT; ++t) {
        #pragma unroll
        for (int r = 0; r < 8; ++r)
            C[(size_t)(m0 + r + 8 * hl) * N + n0 + t * 16 + rr] = acc[t][r];
    }
}

// ---------------------------------------------------------------------------
// Causal depthwise conv (k=4) + bias + SiLU along L.  Thread per (b, ch).
// ---------------------------------------------------------------------------
__global__ void conv_silu_kernel(const float* __restrict__ xz,
                                 const float* __restrict__ cw,
                                 const float* __restrict__ cb,
                                 float* __restrict__ u, bf16n* __restrict__ ub) {
    const int idx = blockIdx.x * blockDim.x + threadIdx.x;
    if (idx >= BATCH * D_INNER) return;
    const int ch = idx & (D_INNER - 1);
    const int b  = idx >> 9;
    const float w0 = cw[ch * 4 + 0], w1 = cw[ch * 4 + 1];
    const float w2 = cw[ch * 4 + 2], w3 = cw[ch * 4 + 3];
    const float bias = cb[ch];
    float x0 = 0.f, x1 = 0.f, x2 = 0.f;
    for (int l = 0; l < L_EFF; ++l) {
        const size_t tok = (size_t)b * L_EFF + l;
        const float x3 = xz[tok * (2 * D_INNER) + ch];       // u half of xz
        float s = fmaf(w0, x0, fmaf(w1, x1, fmaf(w2, x2, fmaf(w3, x3, bias))));
        const float sv = s * (1.f / (1.f + __expf(-s)));     // SiLU
        u [tok * D_INNER + ch] = sv;
        ub[tok * D_INNER + ch] = (bf16n)sv;
        x0 = x1; x1 = x2; x2 = x3;
    }
}

// ---------------------------------------------------------------------------
// dt = softplus(dbc[:, :16] @ dpw^T + dpb).  K=16 -> plain VALU GEMV.
// ---------------------------------------------------------------------------
__global__ void dt_kernel(const float* __restrict__ dbc,
                          const float* __restrict__ dpw,
                          const float* __restrict__ dpb,
                          float* __restrict__ dt) {
    const int idx = blockIdx.x * blockDim.x + threadIdx.x;
    if (idx >= NTOK * D_INNER) return;
    const int n   = idx & (D_INNER - 1);
    const int tok = idx >> 9;
    const float* din = dbc + (size_t)tok * 48;
    float s = dpb[n];
    #pragma unroll
    for (int k = 0; k < DT_RANK; ++k) s = fmaf(din[k], dpw[n * DT_RANK + k], s);
    dt[(size_t)tok * D_INNER + n] = fmaxf(s, 0.f) + log1pf(__expf(-fabsf(s)));
}

// ---------------------------------------------------------------------------
// Selective scan: thread per (b, ch), h[16] in VGPRs, forward over L_EFF.
// Fuses +u*D_skip and *silu(z); emits bf16 y for the out_proj WMMA.
// ---------------------------------------------------------------------------
__global__ void scan_kernel(const float* __restrict__ dt,
                            const float* __restrict__ dbc,
                            const float* __restrict__ u,
                            const float* __restrict__ xz,
                            const float* __restrict__ a_log,
                            const float* __restrict__ d_skip,
                            bf16n* __restrict__ yb) {
    const int idx = blockIdx.x * blockDim.x + threadIdx.x;
    if (idx >= BATCH * D_INNER) return;
    const int ch = idx & (D_INNER - 1);
    const int b  = idx >> 9;
    float An[D_STATE];
    #pragma unroll
    for (int n = 0; n < D_STATE; ++n) An[n] = -__expf(a_log[ch * D_STATE + n]);
    const float dsk = d_skip[ch];
    float h[D_STATE];
    #pragma unroll
    for (int n = 0; n < D_STATE; ++n) h[n] = 0.f;

    for (int l = 0; l < L_EFF; ++l) {
        const size_t tok = (size_t)b * L_EFF + l;
        const float dtv = dt[tok * D_INNER + ch];
        const float uv  = u [tok * D_INNER + ch];
        const float du  = dtv * uv;
        const float* Bc = dbc + tok * 48 + DT_RANK;
        const float* Cc = Bc + D_STATE;
        float y = 0.f;
        #pragma unroll
        for (int n = 0; n < D_STATE; ++n) {
            h[n] = __expf(dtv * An[n]) * h[n] + du * Bc[n];
            y = fmaf(h[n], Cc[n], y);
        }
        y = fmaf(uv, dsk, y);
        const float zv = xz[tok * (2 * D_INNER) + D_INNER + ch];
        y *= zv * (1.f / (1.f + __expf(-zv)));                 // * silu(z)
        yb[tok * D_INNER + ch] = (bf16n)y;
    }
}

// ---------------------------------------------------------------------------
// feat = LayerNorm(yproj + feat).  Block (256 threads) per token.
// ---------------------------------------------------------------------------
__global__ void add_ln_kernel(const float* __restrict__ yproj,
                              const float* __restrict__ g, const float* __restrict__ bt,
                              float* __restrict__ feat, bf16n* __restrict__ featb) {
    __shared__ float red[256];
    const int tok = blockIdx.x, t = threadIdx.x;
    const float v = yproj[(size_t)tok * D_MODEL + t] + feat[(size_t)tok * D_MODEL + t];
    red[t] = v; __syncthreads();
    for (int st = 128; st > 0; st >>= 1) { if (t < st) red[t] += red[t + st]; __syncthreads(); }
    const float mu = red[0] * (1.f / D_MODEL); __syncthreads();
    const float d = v - mu;
    red[t] = d * d; __syncthreads();
    for (int st = 128; st > 0; st >>= 1) { if (t < st) red[t] += red[t + st]; __syncthreads(); }
    const float var = red[0] * (1.f / D_MODEL);
    const float o = d * rsqrtf(var + LN_EPS) * g[t] + bt[t];
    feat [(size_t)tok * D_MODEL + t] = o;
    featb[(size_t)tok * D_MODEL + t] = (bf16n)o;
}

// ---------------------------------------------------------------------------
// Classifier head at position EXIT_POS-1: relu(h@W1^T+b1)@W2^T+b2.
// Block (128 threads) per batch element.
// ---------------------------------------------------------------------------
__global__ void head_kernel(const float* __restrict__ feat,
                            const float* __restrict__ w1, const float* __restrict__ b1,
                            const float* __restrict__ w2, const float* __restrict__ b2,
                            float* __restrict__ out) {
    __shared__ float hid[128];
    const int b = blockIdx.x, t = threadIdx.x;
    const float* h = feat + (size_t)(b * L_EFF + (L_EFF - 1)) * D_MODEL;
    float s = b1[t];
    #pragma unroll 8
    for (int k = 0; k < D_MODEL; ++k) s = fmaf(h[k], w1[t * D_MODEL + k], s);
    hid[t] = fmaxf(s, 0.f);
    __syncthreads();
    if (t < 2) {
        float s2 = b2[t];
        #pragma unroll 8
        for (int k = 0; k < 128; ++k) s2 = fmaf(hid[k], w2[t * 128 + k], s2);
        out[b * 2 + t] = s2;
    }
}

// ---------------------------------------------------------------------------
// Host orchestration
// ---------------------------------------------------------------------------
extern "C" void kernel_launch(void* const* d_in, const int* in_sizes, int n_in,
                              void* d_out, int out_size, void* d_ws, size_t ws_size,
                              hipStream_t stream) {
    (void)in_sizes; (void)n_in; (void)out_size; (void)ws_size;
    const float* x         = (const float*)d_in[0];
    const float* emb_proto = (const float*)d_in[1];
    const float* emb_flags = (const float*)d_in[2];
    const float* emb_dir   = (const float*)d_in[3];
    const float* plw       = (const float*)d_in[4];
    const float* plb       = (const float*)d_in[5];
    const float* piw       = (const float*)d_in[6];
    const float* pib       = (const float*)d_in[7];
    const float* fw        = (const float*)d_in[8];
    const float* fb        = (const float*)d_in[9];
    const float* tng       = (const float*)d_in[10];
    const float* tnb       = (const float*)d_in[11];
    const float* ipw       = (const float*)d_in[12];  // (4,1024,256)
    const float* cw        = (const float*)d_in[13];  // (4,512,4)
    const float* cb        = (const float*)d_in[14];  // (4,512)
    const float* xpw       = (const float*)d_in[15];  // (4,48,512)
    const float* dpw       = (const float*)d_in[16];  // (4,512,16)
    const float* dpb       = (const float*)d_in[17];  // (4,512)
    const float* a_log     = (const float*)d_in[18];  // (4,512,16)
    const float* d_skip    = (const float*)d_in[19];  // (4,512)
    const float* opw       = (const float*)d_in[20];  // (4,256,512)
    const float* ng        = (const float*)d_in[21];
    const float* nb        = (const float*)d_in[22];
    const float* cw1       = (const float*)d_in[23];
    const float* cb1       = (const float*)d_in[24];
    const float* cw2       = (const float*)d_in[25];
    const float* cb2       = (const float*)d_in[26];

    // Workspace carve-up (256B aligned)
    char* ws = (char*)d_ws;
    size_t off = 0;
    auto carve = [&](size_t bytes) -> void* {
        void* p = ws + off;
        off += (bytes + 255) & ~(size_t)255;
        return p;
    };
    float* feat   = (float*)carve((size_t)NTOK * D_MODEL * 4);
    bf16n* featb  = (bf16n*)carve((size_t)NTOK * D_MODEL * 2);
    float* xz     = (float*)carve((size_t)NTOK * 2 * D_INNER * 4);
    float* ubuf   = (float*)carve((size_t)NTOK * D_INNER * 4);
    bf16n* ub     = (bf16n*)carve((size_t)NTOK * D_INNER * 2);
    float* dbc    = (float*)carve((size_t)NTOK * 48 * 4);
    float* dtb    = (float*)carve((size_t)NTOK * D_INNER * 4);
    bf16n* yb     = (bf16n*)carve((size_t)NTOK * D_INNER * 2);
    float* yproj  = (float*)carve((size_t)NTOK * D_MODEL * 4);
    bf16n* wip    = (bf16n*)carve((size_t)N_LAYERS * 1024 * 256 * 2);
    bf16n* wxp    = (bf16n*)carve((size_t)N_LAYERS * 48 * 512 * 2);
    bf16n* wop    = (bf16n*)carve((size_t)N_LAYERS * 256 * 512 * 2);

    // One-time (per call) bf16 weight conversion
    {
        int n1 = N_LAYERS * 1024 * 256;
        int n2 = N_LAYERS * 48 * 512;
        int n3 = N_LAYERS * 256 * 512;
        f2bf_kernel<<<(n1 + 255) / 256, 256, 0, stream>>>(ipw, wip, n1);
        f2bf_kernel<<<(n2 + 255) / 256, 256, 0, stream>>>(xpw, wxp, n2);
        f2bf_kernel<<<(n3 + 255) / 256, 256, 0, stream>>>(opw, wop, n3);
    }

    // Tokenizer + LN
    tok_kernel<<<NTOK, 256, 0, stream>>>(x, emb_proto, emb_flags, emb_dir,
                                         plw, plb, piw, pib, fw, fb, tng, tnb,
                                         feat, featb);

    for (int layer = 0; layer < N_LAYERS; ++layer) {
        const bf16n* wip_l = wip + (size_t)layer * 1024 * 256;
        const bf16n* wxp_l = wxp + (size_t)layer * 48 * 512;
        const bf16n* wop_l = wop + (size_t)layer * 256 * 512;

        // xz = feat @ in_proj^T : (4096 x 1024), K=256, 4 N-tiles per wave
        gemm_bf16_kernel<4><<<dim3(1024 / 64, NTOK / 16), 32, 0, stream>>>(
            featb, wip_l, xz, NTOK, 2 * D_INNER, D_MODEL);

        // causal depthwise conv + SiLU
        conv_silu_kernel<<<(BATCH * D_INNER) / 256, 256, 0, stream>>>(
            xz, cw + (size_t)layer * D_INNER * D_CONV,
            cb + (size_t)layer * D_INNER, ubuf, ub);

        // dbc = u @ x_proj^T : (4096 x 48), K=512, 3 N-tiles per wave
        gemm_bf16_kernel<3><<<dim3(1, NTOK / 16), 32, 0, stream>>>(
            ub, wxp_l, dbc, NTOK, 48, D_INNER);

        // dt = softplus(dt_in @ dt_proj^T + b)
        dt_kernel<<<(NTOK * D_INNER) / 256, 256, 0, stream>>>(
            dbc, dpw + (size_t)layer * D_INNER * DT_RANK,
            dpb + (size_t)layer * D_INNER, dtb);

        // selective scan (+ D_skip + gate)
        scan_kernel<<<(BATCH * D_INNER) / 256, 256, 0, stream>>>(
            dtb, dbc, ubuf, xz,
            a_log + (size_t)layer * D_INNER * D_STATE,
            d_skip + (size_t)layer * D_INNER, yb);

        // yproj = y @ out_proj^T : (4096 x 256), K=512, 4 N-tiles per wave
        gemm_bf16_kernel<4><<<dim3(256 / 64, NTOK / 16), 32, 0, stream>>>(
            yb, wop_l, yproj, NTOK, D_MODEL, D_INNER);

        // feat = LN(yproj + feat)
        add_ln_kernel<<<NTOK, 256, 0, stream>>>(yproj, ng, nb, feat, featb);
    }

    // Head at position 31
    head_kernel<<<BATCH, 128, 0, stream>>>(feat, cw1, cb1, cw2, cb2, (float*)d_out);
}
